// OptimizedSwinBlock_13013750907077
// MI455X (gfx1250) — matrix operand order repaired and checked by reading, not compiled, not run on hardware
//
#include <hip/hip_runtime.h>
#include <hip/hip_bf16.h>
#include <math.h>

typedef __attribute__((ext_vector_type(16))) _Float16 v16h;
typedef __attribute__((ext_vector_type(8)))  float    v8f;

#define C_DIM 96
#define QKV_N 288
#define H1_N  384
#define NT    49     // tokens per 7x7 window
#define MTP   64     // padded rows (multiple of 16)
#define EPSF  1e-5f
#define SCL   0.10206207261596575f  // 1/sqrt(96)

// ---- WMMA fragment helpers (A/B: contiguous K order k = 16*half + s in both; C/D layout per CDNA5 ISA §7.12.2) ----

// A-matrix 16x32 f16: lane<16 -> row=lane, VGPR j holds K = 2j (+16 for hi half)
__device__ __forceinline__ v16h load_A_lds(const _Float16* base, int ld, int m0, int k0, int lane) {
  const int half = lane >> 4, r = lane & 15;
  const _Float16* p = base + (m0 + r) * ld + k0;
  v16h a;
#pragma unroll
  for (int j = 0; j < 8; ++j) {
    int k = 2 * j + half * 16;
    a[2 * j]     = p[k];
    a[2 * j + 1] = p[k + 1];
  }
  return a;
}

// B-matrix 32x16 f16, B[k][n] = base[(k0+k)*ld + n0+n]; VGPR j: K=2j (+16 hi half), N=lane&15
__device__ __forceinline__ v16h load_B_lds(const _Float16* base, int ld, int k0, int n0, int lane) {
  const int half = lane >> 4, n = n0 + (lane & 15);
  v16h b;
#pragma unroll
  for (int j = 0; j < 8; ++j) {
    int k = k0 + 2 * j + half * 16;
    b[2 * j]     = base[k * ld + n];
    b[2 * j + 1] = base[(k + 1) * ld + n];
  }
  return b;
}

// Transposed source (for Q@K^T): B[k][n] = base[(n0+n)*ld + k0+k]; K-pairs are contiguous
__device__ __forceinline__ v16h load_Bt_lds(const _Float16* base, int ld, int k0, int n0, int lane) {
  const int half = lane >> 4, n = n0 + (lane & 15);
  const _Float16* p = base + n * ld + k0 + half * 16;
  v16h b;
#pragma unroll
  for (int j = 0; j < 8; ++j) {
    b[2 * j]     = p[2 * j];
    b[2 * j + 1] = p[2 * j + 1];
  }
  return b;
}

// C/D f32 tile -> LDS f16: VGPR r holds row r (+8 for hi half), col = lane&15
__device__ __forceinline__ void store_C_f16(_Float16* dst, int ld, int m0, int n0, int lane, v8f acc) {
  const int half = lane >> 4, n = n0 + (lane & 15);
#pragma unroll
  for (int r = 0; r < 8; ++r)
    dst[(m0 + r + 8 * half) * ld + n] = (_Float16)acc[r];
}

// ---- weight pre-pack: fp32 row-major W[K][N] -> per-fragment f16 blobs (512 halves) ----
__global__ void pack_b_frags(const float* __restrict__ W, int K, int N, int ktiles,
                             _Float16* __restrict__ dst) {
  const int frag = blockIdx.x;
  const int nt = frag / ktiles, kt = frag % ktiles;
  const int lane = threadIdx.x;
  const int half = lane >> 4, n = nt * 16 + (lane & 15);
  _Float16* out = dst + (size_t)frag * 512 + lane * 16;
  (void)K;
#pragma unroll
  for (int j = 0; j < 8; ++j) {
    int k = kt * 32 + 2 * j + half * 16;
    out[2 * j]     = (_Float16)W[k * N + n];
    out[2 * j + 1] = (_Float16)W[(k + 1) * N + n];
  }
}

// ---- fused LayerNorm + window QKV + attention + window-reverse, 1 workgroup / window ----
__global__ __launch_bounds__(256) void swin_attn_kernel(
    const float* __restrict__ x, const float* __restrict__ ln_w, const float* __restrict__ ln_b,
    const _Float16* __restrict__ qkvw_pk, float* __restrict__ out) {
  extern __shared__ char smem[];
  float*    Of   = (float*)smem;                                        // 64*96 f32: raw-X stage, later O
  float*    Sf   = (float*)(smem + 24576);                              // 64*64 f32: scores
  _Float16* Xs   = (_Float16*)(smem + 24576 + 16384);                   // 64*96 f16: LN'd X
  _Float16* QKVs = (_Float16*)(smem + 24576 + 16384 + 12288);           // 64*288 f16
  _Float16* Pf   = (_Float16*)(smem + 24576 + 16384 + 12288 + 36864);   // 64*64 f16: probs

  const int tid  = threadIdx.x;
  const int wave = tid >> 5, lane = tid & 31;
  const int mt   = wave & 3;        // fixed M-tile per wave
  const int ngrp = wave >> 2;       // N-range selector (0/1)
  const int win = blockIdx.x;       // 8192 windows: b*1024 + wy*32 + wx
  const int bb = win >> 10, rem = win & 1023;
  const int y0 = (rem >> 5) * 7, x0 = (rem & 31) * 7;

  // stage raw 49x96 window into Of as float4; zero the 15 pad rows of Xs
  for (int idx4 = tid; idx4 < NT * (C_DIM / 4); idx4 += 256) {
    int p = idx4 / (C_DIM / 4), c4 = idx4 - p * (C_DIM / 4);
    int yy = y0 + p / 7, xx = x0 + p % 7;
    const float4* src = (const float4*)(x + (((size_t)bb * 224 + yy) * 224 + xx) * C_DIM);
    ((float4*)Of)[idx4] = src[c4];
  }
  for (int idx = tid; idx < (MTP - NT) * C_DIM; idx += 256)
    Xs[NT * C_DIM + idx] = (_Float16)0.f;
  __syncthreads();

  // fused LayerNorm (wave32 shuffle reduction, 3 ch/lane) -> Xs f16
  for (int row = wave; row < NT; row += 8) {
    float v0 = Of[row * C_DIM + lane];
    float v1 = Of[row * C_DIM + lane + 32];
    float v2 = Of[row * C_DIM + lane + 64];
    float s = v0 + v1 + v2;
#pragma unroll
    for (int off = 16; off > 0; off >>= 1) s += __shfl_xor(s, off, 32);
    float mu = s * (1.f / 96.f);
    float d0 = v0 - mu, d1 = v1 - mu, d2 = v2 - mu;
    float q = d0 * d0 + d1 * d1 + d2 * d2;
#pragma unroll
    for (int off = 16; off > 0; off >>= 1) q += __shfl_xor(q, off, 32);
    float rs = rsqrtf(q * (1.f / 96.f) + EPSF);
    Xs[row * C_DIM + lane]      = (_Float16)(d0 * rs * ln_w[lane]      + ln_b[lane]);
    Xs[row * C_DIM + lane + 32] = (_Float16)(d1 * rs * ln_w[lane + 32] + ln_b[lane + 32]);
    Xs[row * C_DIM + lane + 64] = (_Float16)(d2 * rs * ln_w[lane + 64] + ln_b[lane + 64]);
  }
  __syncthreads();

  // QKV = Xs(64x96) @ Wqkv(96x288). A-frags loaded once, reused over 9 N-tiles.
  {
    v16h a0 = load_A_lds(Xs, C_DIM, mt * 16, 0,  lane);
    v16h a1 = load_A_lds(Xs, C_DIM, mt * 16, 32, lane);
    v16h a2 = load_A_lds(Xs, C_DIM, mt * 16, 64, lane);
    for (int nt = ngrp * 9; nt < ngrp * 9 + 9; ++nt) {
      const _Float16* wp = qkvw_pk + (size_t)(nt * 3) * 512 + lane * 16;
      v8f acc = {};
      acc = __builtin_amdgcn_wmma_f32_16x16x32_f16(false, a0, false, *(const v16h*)(wp),        (short)0, acc, false, false);
      acc = __builtin_amdgcn_wmma_f32_16x16x32_f16(false, a1, false, *(const v16h*)(wp + 512),  (short)0, acc, false, false);
      acc = __builtin_amdgcn_wmma_f32_16x16x32_f16(false, a2, false, *(const v16h*)(wp + 1024), (short)0, acc, false, false);
      store_C_f16(QKVs, QKV_N, mt * 16, nt * 16, lane, acc);
    }
  }
  __syncthreads();

  // S = (Q @ K^T) * 1/sqrt(96). Q-frags reused over 2 N-tiles per wave.
  {
    v16h a0 = load_A_lds(QKVs, QKV_N, mt * 16, 0,  lane);
    v16h a1 = load_A_lds(QKVs, QKV_N, mt * 16, 32, lane);
    v16h a2 = load_A_lds(QKVs, QKV_N, mt * 16, 64, lane);
    const int half = lane >> 4, nn = lane & 15;
#pragma unroll
    for (int i = 0; i < 2; ++i) {
      int nt = ngrp * 2 + i;
      v8f acc = {};
      acc = __builtin_amdgcn_wmma_f32_16x16x32_f16(false, a0, false, load_Bt_lds(QKVs + 96, QKV_N, 0,  nt * 16, lane), (short)0, acc, false, false);
      acc = __builtin_amdgcn_wmma_f32_16x16x32_f16(false, a1, false, load_Bt_lds(QKVs + 96, QKV_N, 32, nt * 16, lane), (short)0, acc, false, false);
      acc = __builtin_amdgcn_wmma_f32_16x16x32_f16(false, a2, false, load_Bt_lds(QKVs + 96, QKV_N, 64, nt * 16, lane), (short)0, acc, false, false);
#pragma unroll
      for (int r = 0; r < 8; ++r)
        Sf[(mt * 16 + r + 8 * half) * MTP + nt * 16 + nn] = acc[r] * SCL;
    }
  }
  __syncthreads();

  // masked softmax: 4 threads/row, 16 cols each, 2-step shuffle reduce in the 4-lane group
  {
    int row = tid >> 2, q = tid & 3;
    if (row < NT) {
      float e[16];
      float m = -3.4e38f;
#pragma unroll
      for (int j = 0; j < 16; ++j) {
        int col = q * 16 + j;
        float v = (col < NT) ? Sf[row * MTP + col] : -3.4e38f;
        e[j] = v;
        m = fmaxf(m, v);
      }
      m = fmaxf(m, __shfl_xor(m, 1, 32));
      m = fmaxf(m, __shfl_xor(m, 2, 32));
      float s = 0.f;
#pragma unroll
      for (int j = 0; j < 16; ++j) {
        int col = q * 16 + j;
        e[j] = (col < NT) ? __expf(e[j] - m) : 0.f;
        s += e[j];
      }
      s += __shfl_xor(s, 1, 32);
      s += __shfl_xor(s, 2, 32);
      float inv = 1.f / s;
#pragma unroll
      for (int j = 0; j < 16; ++j)
        Pf[row * MTP + q * 16 + j] = (_Float16)(e[j] * inv);
    } else {
#pragma unroll
      for (int j = 0; j < 16; ++j)
        Pf[row * MTP + q * 16 + j] = (_Float16)0.f;
    }
  }
  __syncthreads();

  // O = P(64x64) @ V(64x96). P-frags reused over 3 N-tiles per wave.
  {
    v16h a0 = load_A_lds(Pf, MTP, mt * 16, 0,  lane);
    v16h a1 = load_A_lds(Pf, MTP, mt * 16, 32, lane);
    const int half = lane >> 4, nn = lane & 15;
#pragma unroll
    for (int i = 0; i < 3; ++i) {
      int nt = ngrp * 3 + i;
      v8f acc = {};
      acc = __builtin_amdgcn_wmma_f32_16x16x32_f16(false, a0, false, load_B_lds(QKVs + 192, QKV_N, 0,  nt * 16, lane), (short)0, acc, false, false);
      acc = __builtin_amdgcn_wmma_f32_16x16x32_f16(false, a1, false, load_B_lds(QKVs + 192, QKV_N, 32, nt * 16, lane), (short)0, acc, false, false);
#pragma unroll
      for (int r = 0; r < 8; ++r)
        Of[(mt * 16 + r + 8 * half) * C_DIM + nt * 16 + nn] = acc[r];
    }
  }
  __syncthreads();

  // window-reverse scatter of the 49 valid rows into (b,h,w,c) as float4
  for (int idx4 = tid; idx4 < NT * (C_DIM / 4); idx4 += 256) {
    int p = idx4 / (C_DIM / 4), c4 = idx4 - p * (C_DIM / 4);
    int yy = y0 + p / 7, xx = x0 + p % 7;
    float4* dst = (float4*)(out + (((size_t)bb * 224 + yy) * 224 + xx) * C_DIM);
    dst[c4] = ((const float4*)Of)[idx4];
  }
}

// ---- fused MLP: out = gelu(X@W1+b1)@W2 + b2 + X, 1 workgroup / 64 tokens, in-place on d_out ----
__global__ __launch_bounds__(256) void swin_mlp_kernel(
    const _Float16* __restrict__ w1_pk, const _Float16* __restrict__ w2_pk,
    const float* __restrict__ b1, const float* __restrict__ b2,
    float* __restrict__ io) {
  extern __shared__ char smem[];
  float*    O32 = (float*)smem;                       // 64*96 f32: residual copy
  _Float16* A16 = (_Float16*)(smem + 24576);          // 64*96 f16: GEMM input
  _Float16* H16 = (_Float16*)(smem + 24576 + 12288);  // 64*384 f16: hidden

  const int tid  = threadIdx.x;
  const int wave = tid >> 5, lane = tid & 31;
  const int mt   = wave & 3;
  const int ngrp = wave >> 2;
  const size_t m0 = (size_t)blockIdx.x * 64;

  for (int idx4 = tid; idx4 < 64 * (C_DIM / 4); idx4 += 256) {
    float4 v = ((const float4*)(io + m0 * C_DIM))[idx4];
    ((float4*)O32)[idx4] = v;
    A16[idx4 * 4 + 0] = (_Float16)v.x;
    A16[idx4 * 4 + 1] = (_Float16)v.y;
    A16[idx4 * 4 + 2] = (_Float16)v.z;
    A16[idx4 * 4 + 3] = (_Float16)v.w;
  }
  __syncthreads();

  // H = gelu(A @ W1 + b1). A-frags loaded once, reused over 12 N-tiles per wave.
  {
    v16h a0 = load_A_lds(A16, C_DIM, mt * 16, 0,  lane);
    v16h a1 = load_A_lds(A16, C_DIM, mt * 16, 32, lane);
    v16h a2 = load_A_lds(A16, C_DIM, mt * 16, 64, lane);
    const int half = lane >> 4, nn = lane & 15;
    for (int nt = ngrp * 12; nt < ngrp * 12 + 12; ++nt) {
      const _Float16* wp = w1_pk + (size_t)(nt * 3) * 512 + lane * 16;
      v8f acc = {};
      acc = __builtin_amdgcn_wmma_f32_16x16x32_f16(false, a0, false, *(const v16h*)(wp),        (short)0, acc, false, false);
      acc = __builtin_amdgcn_wmma_f32_16x16x32_f16(false, a1, false, *(const v16h*)(wp + 512),  (short)0, acc, false, false);
      acc = __builtin_amdgcn_wmma_f32_16x16x32_f16(false, a2, false, *(const v16h*)(wp + 1024), (short)0, acc, false, false);
      int n = nt * 16 + nn;
      float bias = b1[n];
#pragma unroll
      for (int r = 0; r < 8; ++r) {
        float v = acc[r] + bias;
        v = 0.5f * v * (1.f + erff(v * 0.70710678118654752f));  // exact gelu
        H16[(mt * 16 + r + 8 * half) * H1_N + n] = (_Float16)v;
      }
    }
  }
  __syncthreads();

  // out = H @ W2 + b2 + O. K outermost: 1 A-frag feeds 3 accumulators (3 N-tiles/wave).
  {
    const int half = lane >> 4, nn = lane & 15;
    v8f acc0 = {}, acc1 = {}, acc2 = {};
    const int ntb = ngrp * 3;
#pragma unroll
    for (int kt = 0; kt < 12; ++kt) {
      v16h a = load_A_lds(H16, H1_N, mt * 16, kt * 32, lane);
      acc0 = __builtin_amdgcn_wmma_f32_16x16x32_f16(false, a, false, *(const v16h*)(w2_pk + (size_t)((ntb + 0) * 12 + kt) * 512 + lane * 16), (short)0, acc0, false, false);
      acc1 = __builtin_amdgcn_wmma_f32_16x16x32_f16(false, a, false, *(const v16h*)(w2_pk + (size_t)((ntb + 1) * 12 + kt) * 512 + lane * 16), (short)0, acc1, false, false);
      acc2 = __builtin_amdgcn_wmma_f32_16x16x32_f16(false, a, false, *(const v16h*)(w2_pk + (size_t)((ntb + 2) * 12 + kt) * 512 + lane * 16), (short)0, acc2, false, false);
    }
#pragma unroll
    for (int i = 0; i < 3; ++i) {
      v8f acc = (i == 0) ? acc0 : (i == 1) ? acc1 : acc2;
      int n = (ntb + i) * 16 + nn;
      float bias = b2[n];
#pragma unroll
      for (int r = 0; r < 8; ++r) {
        int row = mt * 16 + r + 8 * half;
        io[(m0 + row) * C_DIM + n] = acc[r] + bias + O32[row * C_DIM + n];
      }
    }
  }
}

extern "C" void kernel_launch(void* const* d_in, const int* in_sizes, int n_in,
                              void* d_out, int out_size, void* d_ws, size_t ws_size,
                              hipStream_t stream) {
  (void)in_sizes; (void)n_in; (void)out_size; (void)ws_size;
  const float* x     = (const float*)d_in[0];
  const float* qkv_w = (const float*)d_in[1];
  const float* ln_w  = (const float*)d_in[2];
  const float* ln_b  = (const float*)d_in[3];
  const float* w1    = (const float*)d_in[4];
  const float* b1    = (const float*)d_in[5];
  const float* w2    = (const float*)d_in[6];
  const float* b2    = (const float*)d_in[7];
  float* out = (float*)d_out;

  // workspace: fragment-packed f16 weights only (~200 KB, L2-resident)
  _Float16* qkvw_pk = (_Float16*)d_ws;          // 18 N-tiles * 3 K-tiles = 54 frags
  _Float16* w1_pk   = qkvw_pk + 54 * 512;       // 24 * 3 = 72 frags
  _Float16* w2_pk   = w1_pk + 72 * 512;         // 6 * 12 = 72 frags

  pack_b_frags<<<54, 32, 0, stream>>>(qkv_w, 96, 288, 3, qkvw_pk);
  pack_b_frags<<<72, 32, 0, stream>>>(w1,    96, 384, 3, w1_pk);
  pack_b_frags<<<72, 32, 0, stream>>>(w2,   384,  96, 12, w2_pk);

  // 8192 windows: LN + QKV + attention + window-reverse, output -> d_out
  swin_attn_kernel<<<8192, 256, 98304, stream>>>(x, ln_w, ln_b, qkvw_pk, out);
  // 401408/64 = 6272 token tiles: MLP + residual, in-place on d_out
  swin_mlp_kernel<<<6272, 256, 86016, stream>>>(w1_pk, w2_pk, b1, b2, out);
}
